// Postprocessor_30588757082608
// MI455X (gfx1250) — compile-verified, matrix operand order checked by reading, not run on hardware
//
#include <hip/hip_runtime.h>
#include <hip/hip_bf16.h>
#include <stdint.h>

#define NPTS      21760
#define NC        80
#define PRED_ROW  112          // 80 + 4*8
#define PRE_K     4096
#define CONF      0.35f
#define IOU_T     0.6f
#define NMSMAX    100
#define INSZ      1024.0f

typedef _Float16 v16h __attribute__((ext_vector_type(16)));
typedef float    v8f  __attribute__((ext_vector_type(8)));
typedef unsigned v4u  __attribute__((ext_vector_type(4)));
typedef int      v8i  __attribute__((ext_vector_type(8)));
typedef int      v4i  __attribute__((ext_vector_type(4)));

// ---------------------------------------------------------------- init
__global__ void k_init(float* candScore, unsigned* candKey, unsigned* counter) {
    int t = blockIdx.x * blockDim.x + threadIdx.x;
    if (t == 0) *counter = 0u;
    if (t < PRE_K) { candScore[t] = -1.0f; candKey[t] = 0u; }
}

// -------------------------------------------------- DFL decode via WMMA
// Each wave handles 4 points = 16 rows of the (n*4) x 8 softmax-prob matrix.
// B column 0 = bin index (0..7), column 1 = ones -> D col0 = numerator,
// D col1 = denominator of the softmax expected value.
__global__ __launch_bounds__(256) void k_decode(const float* __restrict__ preds,
                                                float* __restrict__ bboxes) {
    __shared__ float nd[8][16][2];
    const int wave = threadIdx.x >> 5;
    const int lane = threadIdx.x & 31;
    const int n0   = (blockIdx.x * 8 + wave) * 4;     // first point of this wave
    if (n0 >= NPTS) return;

    const int row  = lane & 15;                       // M index (0..15)
    const int pt   = n0 + (row >> 2);
    const int side = row & 3;

    // A: 16x32 f16, only K=0..7 populated (lanes 0-15 carry K0..7 per ISA layout)
    v16h a = {};
    if (lane < 16) {
        const float* r = preds + (size_t)pt * PRED_ROW + NC + side * 8;
        float x[8];
#pragma unroll
        for (int j = 0; j < 8; ++j) x[j] = r[j];
        float m = x[0];
#pragma unroll
        for (int j = 1; j < 8; ++j) m = fmaxf(m, x[j]);
#pragma unroll
        for (int j = 0; j < 8; ++j) a[j] = (_Float16)__expf(x[j] - m);
    }
    // B: 32x16 f16, col0 = K (bin index), col1 = 1, else 0
    v16h b = {};
    if (lane == 0) {
#pragma unroll
        for (int j = 0; j < 8; ++j) b[j] = (_Float16)(float)j;
    } else if (lane == 1) {
#pragma unroll
        for (int j = 0; j < 8; ++j) b[j] = (_Float16)1.0f;
    }

    v8f c = {};
    c = __builtin_amdgcn_wmma_f32_16x16x32_f16(false, a, false, b,
                                               (short)0, c, false, false);

    // Scatter D col0/col1: lanes {0,1} hold rows 0..7, lanes {16,17} rows 8..15
    if (lane == 0)  { for (int r2 = 0; r2 < 8; ++r2) nd[wave][r2][0]     = c[r2]; }
    if (lane == 1)  { for (int r2 = 0; r2 < 8; ++r2) nd[wave][r2][1]     = c[r2]; }
    if (lane == 16) { for (int r2 = 0; r2 < 8; ++r2) nd[wave][8 + r2][0] = c[r2]; }
    if (lane == 17) { for (int r2 = 0; r2 < 8; ++r2) nd[wave][8 + r2][1] = c[r2]; }
    __syncthreads();

    if (lane < 4) {
        int n = n0 + lane;
        int s, fs, base;
        if      (n < 16384) { s = 8;  fs = 128; base = 0; }
        else if (n < 20480) { s = 16; fs = 64;  base = 16384; }
        else if (n < 21504) { s = 32; fs = 32;  base = 20480; }
        else                { s = 64; fs = 16;  base = 21504; }
        int cell = n - base;
        float cx = (float)((cell % fs) * s);
        float cy = (float)((cell / fs) * s);
        float d[4];
#pragma unroll
        for (int k = 0; k < 4; ++k) {
            float num = nd[wave][lane * 4 + k][0];
            float den = nd[wave][lane * 4 + k][1];
            d[k] = (num / den) * (float)s;
        }
        bboxes[n * 4 + 0] = fminf(fmaxf(cx - d[0], 0.0f), INSZ);
        bboxes[n * 4 + 1] = fminf(fmaxf(cy - d[1], 0.0f), INSZ);
        bboxes[n * 4 + 2] = fminf(fmaxf(cx + d[2], 0.0f), INSZ);
        bboxes[n * 4 + 3] = fminf(fmaxf(cy + d[3], 0.0f), INSZ);
    }
}

// -------- fused sigmoid + threshold + wave-aggregated compaction (wave32)
__global__ void k_scores(const float* __restrict__ preds,
                         float* __restrict__ candScore,
                         unsigned* __restrict__ candKey,
                         unsigned* __restrict__ counter) {
    int t = blockIdx.x * blockDim.x + threadIdx.x;
    bool pass = false;
    float s = 0.0f;
    if (t < NPTS * NC) {
        int n = t / NC, c = t % NC;
        float v = preds[(size_t)n * PRED_ROW + c];
        s = 1.0f / (1.0f + __expf(-v));
        pass = (s > CONF);
    }
    // one atomic per wave instead of one per passing lane
    unsigned long long m = __ballot(pass);
    if (m != 0ull) {
        int lane   = threadIdx.x & 31;
        int leader = (int)(__ffsll((long long)m) - 1);
        unsigned base = 0u;
        if (lane == leader) base = atomicAdd(counter, (unsigned)__popcll(m));
        base = __shfl(base, leader, 32);
        if (pass) {
            unsigned rank = (unsigned)__popcll(m & ((1ull << lane) - 1ull));
            unsigned pos  = base + rank;
            if (pos < PRE_K) { candScore[pos] = s; candKey[pos] = (unsigned)t; }
        }
    }
}

// ---------------------- bitonic sort (descending, key tiebreak) in LDS
__global__ __launch_bounds__(1024) void k_sort(float* candScore, unsigned* candKey) {
    __shared__ float    ss[PRE_K];
    __shared__ unsigned kk[PRE_K];
    const int tid = threadIdx.x;
    for (int i = tid; i < PRE_K; i += 1024) { ss[i] = candScore[i]; kk[i] = candKey[i]; }
    __syncthreads();
    for (int size = 2; size <= PRE_K; size <<= 1) {
        for (int stride = size >> 1; stride > 0; stride >>= 1) {
            for (int idx = tid; idx < PRE_K; idx += 1024) {
                if ((idx & stride) == 0) {
                    int p = idx | stride;
                    bool wantDesc = ((idx & size) == 0);
                    bool lt = (ss[idx] < ss[p]) ||
                              (ss[idx] == ss[p] && kk[idx] > kk[p]);
                    if (lt == wantDesc) {
                        float tf = ss[idx]; ss[idx] = ss[p]; ss[p] = tf;
                        unsigned tk = kk[idx]; kk[idx] = kk[p]; kk[p] = tk;
                    }
                }
            }
            __syncthreads();
        }
    }
    for (int i = tid; i < PRE_K; i += 1024) { candScore[i] = ss[i]; candKey[i] = kk[i]; }
}

// --------------- gather candidate boxes, apply class offset for IoU
__global__ void k_prep(const float* __restrict__ candScore,
                       const unsigned* __restrict__ candKey,
                       const float* __restrict__ bboxes,
                       float* __restrict__ cbox,      // class-offset boxes
                       float* __restrict__ rawbox,
                       unsigned* __restrict__ cls) {
    int i = blockIdx.x * blockDim.x + threadIdx.x;
    if (i >= PRE_K) return;
    float s = candScore[i];
    if (s > CONF) {
        unsigned key = candKey[i];
        unsigned n = key / NC, c = key % NC;
        float off = (float)c * (INSZ + 1.0f);
#pragma unroll
        for (int j = 0; j < 4; ++j) {
            float b = bboxes[n * 4 + j];
            rawbox[i * 4 + j] = b;
            cbox[i * 4 + j]   = b + off;
        }
        cls[i] = c;
    } else {
#pragma unroll
        for (int j = 0; j < 4; ++j) { rawbox[i * 4 + j] = 0.0f; cbox[i * 4 + j] = 0.0f; }
        cls[i] = 0xFFFFFFFFu;
    }
}

// ------------------- suppression bitmask: sup[i][w] bits for j>i, iou>thr
// j-boxes staged into LDS via Tensor Data Mover (32KB tile).
#define JTILE 2048
__global__ __launch_bounds__(256) void k_nms_matrix(const float* __restrict__ cbox,
                                                    unsigned* __restrict__ sup) {
    __shared__ float jb[JTILE * 4];                 // 32 KB
    const int i0 = blockIdx.x * 64;

    for (int jt = 0; jt < 2; ++jt) {
#if __has_builtin(__builtin_amdgcn_tensor_load_to_lds) && __has_builtin(__builtin_amdgcn_s_wait_tensorcnt)
        if (threadIdx.x == 0) {
            // ---- pack 1D tensor descriptor (D#) per CDNA5 ISA 8.3/8.4 ----
            unsigned long long ga =
                (unsigned long long)(uintptr_t)(cbox + (size_t)jt * JTILE * 4);
            unsigned ldsOff = (unsigned)(uintptr_t)(void*)&jb[0]; // low 32 = LDS byte addr
            const unsigned tensorElems = PRE_K * 4;   // whole cbox array, 4B units
            const unsigned tileElems   = JTILE * 4;   // 8192 elements = 32KB
            v4u g0;
            g0[0] = 1u;                                           // count=1, user desc
            g0[1] = ldsOff;                                       // lds_addr
            g0[2] = (unsigned)(ga & 0xFFFFFFFFu);                 // global_addr[31:0]
            g0[3] = (unsigned)((ga >> 32) & 0x1FFFFFFu) | (2u << 30); // [56:32] | type=2
            v8i g1;
            g1[0] = (int)(2u << 16);                              // data_size=4B, mask=0
            g1[1] = (int)((tensorElems & 0xFFFFu) << 16);         // tensor_dim0 lo
            g1[2] = (int)(((tensorElems >> 16) & 0xFFFFu) | (1u << 16)); // dim0 hi | dim1=1
            g1[3] = (int)((tileElems & 0xFFFFu) << 16);           // dim1 hi=0 | tile_dim0
            g1[4] = 0;                                            // tile_dim1/2 unused
            g1[5] = (int)tensorElems;                             // dim0_stride lo
            g1[6] = 0;                                            // stride hi | dim1_stride lo
            g1[7] = 0;
            v4i g2 = {0, 0, 0, 0};
            v4i g3 = {0, 0, 0, 0};
#if defined(__clang_major__) && (__clang_major__ >= 23)
            v8i g4 = {0, 0, 0, 0, 0, 0, 0, 0};
            __builtin_amdgcn_tensor_load_to_lds(g0, g1, g2, g3, g4, 0);
#else
            __builtin_amdgcn_tensor_load_to_lds(g0, g1, g2, g3, 0);
#endif
            __builtin_amdgcn_s_wait_tensorcnt(0);
        }
#else
        for (int e = threadIdx.x; e < JTILE * 4; e += 256)
            jb[e] = cbox[(size_t)jt * JTILE * 4 + e];
#endif
        __syncthreads();

        for (int task = threadIdx.x; task < 64 * (JTILE / 32); task += 256) {
            int il = task >> 6;                 // local i (0..63)
            int w  = task & 63;                 // word within tile
            int i  = i0 + il;
            float ax1 = cbox[i * 4 + 0], ay1 = cbox[i * 4 + 1];
            float ax2 = cbox[i * 4 + 2], ay2 = cbox[i * 4 + 3];
            float areaA = (ax2 - ax1) * (ay2 - ay1);
            unsigned m = 0u;
            int jb0 = w * 32;
#pragma unroll 4
            for (int jj = 0; jj < 32; ++jj) {
                int j = jt * JTILE + jb0 + jj;
                if (j <= i) continue;                      // reference: idx > i
                float bx1 = jb[(jb0 + jj) * 4 + 0], by1 = jb[(jb0 + jj) * 4 + 1];
                float bx2 = jb[(jb0 + jj) * 4 + 2], by2 = jb[(jb0 + jj) * 4 + 3];
                float areaB = (bx2 - bx1) * (by2 - by1);
                float ix1 = fmaxf(ax1, bx1), iy1 = fmaxf(ay1, by1);
                float ix2 = fminf(ax2, bx2), iy2 = fminf(ay2, by2);
                float iw = fmaxf(ix2 - ix1, 0.0f), ih = fmaxf(iy2 - iy1, 0.0f);
                float inter = iw * ih;
                float iou = inter / (areaA + areaB - inter + 1e-6f);
                if (iou > IOU_T) m |= (1u << jj);
            }
            sup[(size_t)i * 128 + jt * 64 + w] = m;
        }
        __syncthreads();
    }
}

// --------------- greedy serial scan, single wave, keep mask in 4 VGPRs/lane
__global__ __launch_bounds__(32) void k_nms_scan(const float* __restrict__ candScore,
                                                 const unsigned* __restrict__ sup,
                                                 unsigned* __restrict__ keepw) {
    const int lane = threadIdx.x;          // 0..31, owns words 4*lane .. 4*lane+3
    unsigned kw[4];
#pragma unroll
    for (int q = 0; q < 4; ++q) {
        int word = lane * 4 + q;
        unsigned m = 0u;
        for (int b = 0; b < 32; ++b)
            if (candScore[word * 32 + b] > CONF) m |= (1u << b);
        kw[q] = m;
    }
    for (int i = 0; i < PRE_K; ++i) {
        // hide suppression-row latency: warm L2/L0 eight rows ahead
        if (i + 8 < PRE_K)
            __builtin_prefetch(sup + (size_t)(i + 8) * 128 + lane * 4, 0, 1);
        int srcLane = i >> 7;
        int q = (i >> 5) & 3;
        unsigned sel;
        switch (q) {
            case 0:  sel = kw[0]; break;
            case 1:  sel = kw[1]; break;
            case 2:  sel = kw[2]; break;
            default: sel = kw[3]; break;
        }
        unsigned wv = __shfl(sel, srcLane, 32);
        if ((wv >> (i & 31)) & 1u) {       // uniform branch: i survives
#pragma unroll
            for (int q2 = 0; q2 < 4; ++q2)
                kw[q2] &= ~sup[(size_t)i * 128 + lane * 4 + q2];
        }
    }
#pragma unroll
    for (int q = 0; q < 4; ++q) keepw[lane * 4 + q] = kw[q];
}

// --------------- top-100 compact + inverse warp + clip + write output
__global__ __launch_bounds__(128) void k_final(const float* __restrict__ candScore,
                                               const unsigned* __restrict__ cls,
                                               const float* __restrict__ rawbox,
                                               const unsigned* __restrict__ keepw,
                                               const float* __restrict__ warp,
                                               const int* __restrict__ hptr,
                                               const int* __restrict__ wptr,
                                               float* __restrict__ out) {
    __shared__ int selIdx[NMSMAX];
    __shared__ int selCount;
    if (threadIdx.x == 0) {
        int cnt = 0;
        for (int i = 0; i < PRE_K && cnt < NMSMAX; ++i)
            if ((keepw[i >> 5] >> (i & 31)) & 1u) selIdx[cnt++] = i;
        selCount = cnt;
    }
    __syncthreads();
    int t = threadIdx.x;
    if (t >= NMSMAX) return;

    if (t < selCount) {
        int i = selIdx[t];
        float score = candScore[i];
        float x1 = rawbox[i * 4 + 0], y1 = rawbox[i * 4 + 1];
        float x2 = rawbox[i * 4 + 2], y2 = rawbox[i * 4 + 3];
        // 3x3 inverse of warp (adjugate / det)
        float m0 = warp[0], m1 = warp[1], m2 = warp[2];
        float m3 = warp[3], m4 = warp[4], m5 = warp[5];
        float m6 = warp[6], m7 = warp[7], m8 = warp[8];
        float det = m0 * (m4 * m8 - m5 * m7) - m1 * (m3 * m8 - m5 * m6)
                  + m2 * (m3 * m7 - m4 * m6);
        float id = 1.0f / det;
        float i0 = (m4 * m8 - m5 * m7) * id, i1 = (m2 * m7 - m1 * m8) * id,
              i2 = (m1 * m5 - m2 * m4) * id;
        float i3 = (m5 * m6 - m3 * m8) * id, i4 = (m0 * m8 - m2 * m6) * id,
              i5 = (m2 * m3 - m0 * m5) * id;
        float i6 = (m3 * m7 - m4 * m6) * id, i7 = (m1 * m6 - m0 * m7) * id,
              i8 = (m0 * m4 - m1 * m3) * id;
        float cxs[4] = {x1, x2, x2, x1};
        float cys[4] = {y1, y1, y2, y2};
        float lox = 1e30f, loy = 1e30f, hix = -1e30f, hiy = -1e30f;
#pragma unroll
        for (int k = 0; k < 4; ++k) {
            float X = i0 * cxs[k] + i1 * cys[k] + i2;
            float Y = i3 * cxs[k] + i4 * cys[k] + i5;
            float W = i6 * cxs[k] + i7 * cys[k] + i8;
            float px = X / W, py = Y / W;
            lox = fminf(lox, px); hix = fmaxf(hix, px);
            loy = fminf(loy, py); hiy = fmaxf(hiy, py);
        }
        float Wd = (float)(*wptr), Hd = (float)(*hptr);
        out[t * 5 + 0] = fminf(fmaxf(lox, 0.0f), Wd);
        out[t * 5 + 1] = fminf(fmaxf(loy, 0.0f), Hd);
        out[t * 5 + 2] = fminf(fmaxf(hix, 0.0f), Wd);
        out[t * 5 + 3] = fminf(fmaxf(hiy, 0.0f), Hd);
        out[t * 5 + 4] = score;
        out[NMSMAX * 5 + t] = (float)cls[i];
    } else {
#pragma unroll
        for (int j = 0; j < 5; ++j) out[t * 5 + j] = 0.0f;
        out[NMSMAX * 5 + t] = -1.0f;
    }
}

// ---------------------------------------------------------------- launch
extern "C" void kernel_launch(void* const* d_in, const int* in_sizes, int n_in,
                              void* d_out, int out_size, void* d_ws, size_t ws_size,
                              hipStream_t stream) {
    const float* preds = (const float*)d_in[0];
    const float* warp  = (const float*)d_in[2];
    const int*   hptr  = (const int*)d_in[3];
    const int*   wptr  = (const int*)d_in[4];
    float* out = (float*)d_out;

    char* ws = (char*)d_ws;
    float*    bboxes    = (float*)(ws + 0);            // 21760*4*4 = 348160
    unsigned* counter   = (unsigned*)(ws + 348160);    // 256
    float*    candScore = (float*)(ws + 348416);       // 16384
    unsigned* candKey   = (unsigned*)(ws + 364800);    // 16384
    float*    cbox      = (float*)(ws + 381184);       // 65536
    float*    rawbox    = (float*)(ws + 446720);       // 65536
    unsigned* cls       = (unsigned*)(ws + 512256);    // 16384
    unsigned* sup       = (unsigned*)(ws + 528640);    // 4096*128*4 = 2097152
    unsigned* keepw     = (unsigned*)(ws + 2625792);   // 512

    k_init<<<(PRE_K + 255) / 256, 256, 0, stream>>>(candScore, candKey, counter);
    k_decode<<<NPTS / 32, 256, 0, stream>>>(preds, bboxes);
    k_scores<<<(NPTS * NC + 255) / 256, 256, 0, stream>>>(preds, candScore, candKey, counter);
    k_sort<<<1, 1024, 0, stream>>>(candScore, candKey);
    k_prep<<<PRE_K / 256, 256, 0, stream>>>(candScore, candKey, bboxes, cbox, rawbox, cls);
    k_nms_matrix<<<PRE_K / 64, 256, 0, stream>>>(cbox, sup);
    k_nms_scan<<<1, 32, 0, stream>>>(candScore, sup, keepw);
    k_final<<<1, 128, 0, stream>>>(candScore, cls, rawbox, keepw, warp, hptr, wptr, out);
}
